// EntityAttentionLayer_14757507629768
// MI455X (gfx1250) — matrix operand order repaired and checked by reading, not compiled, not run on hardware
//
#include <hip/hip_runtime.h>

typedef __attribute__((ext_vector_type(16))) __bf16 v16bf;
typedef __attribute__((ext_vector_type(8)))  __bf16 v8bf;
typedef __attribute__((ext_vector_type(8)))  float  v8f;
typedef __attribute__((ext_vector_type(4)))  float  v4f;

namespace {
constexpr int kBatch   = 8192;
constexpr int kAgents  = 16;
constexpr int kEnt     = 32;
constexpr int kDim     = 128;
constexpr int kWPitch  = 136;   // 128 + 8 pad -> conflict-free row reads
constexpr int kWoPitch = 264;   // 256 + 8 pad
constexpr int kQaPitch = 136;
constexpr int kKPitch  = 136;
constexpr int kVtPitch = 40;    // 32 + 8 pad
constexpr int kGrid    = 256;
constexpr int kWaves   = 8;
constexpr int kBpw     = kBatch / (kGrid * kWaves); // 4 batches per wave
}

struct WaveScratch {
  alignas(16) __bf16 qa[kAgents * kQaPitch]; // Q, later attended (row-major [a][d])
  alignas(16) __bf16 kv[5120];               // K[32][136] then V^T[128][40]
};
struct Smem {
  alignas(16) __bf16 wq[kDim * kWPitch];
  alignas(16) __bf16 wk[kDim * kWPitch];
  alignas(16) __bf16 wv[kDim * kWPitch];
  alignas(16) __bf16 wo[kDim * kWoPitch];
  WaveScratch ws[kWaves];
};

static __device__ __forceinline__ v16bf cat16(v8bf a, v8bf b) {
  return __builtin_shufflevector(a, b, 0,1,2,3,4,5,6,7,8,9,10,11,12,13,14,15);
}
// B-fragment: 16 contiguous bf16 (two 16B LDS loads)
static __device__ __forceinline__ v16bf ldB(const __bf16* p) {
  return cat16(*(const v8bf*)p, *(const v8bf*)(p + 8));
}
// A-fragment: 8 halves at p, 8 halves at p+16 (K and K+16 slices)
static __device__ __forceinline__ v16bf ldA(const __bf16* p) {
  return cat16(*(const v8bf*)p, *(const v8bf*)(p + 16));
}
static __device__ __forceinline__ v8f wmma_bf16(v16bf a, v16bf b, v8f c) {
  return __builtin_amdgcn_wmma_f32_16x16x32_bf16(false, a, false, b, (short)0, c,
                                                 false, false);
}
static __device__ __forceinline__ v8f vzero8() {
  v8f z = {0.f, 0.f, 0.f, 0.f, 0.f, 0.f, 0.f, 0.f};
  return z;
}

__global__ __launch_bounds__(256, 1) void entity_attention_kernel(
    const float* __restrict__ X, const unsigned char* __restrict__ pre_mask,
    const unsigned char* __restrict__ post_mask,
    const float* __restrict__ Wq, const float* __restrict__ bq,
    const float* __restrict__ Wk, const float* __restrict__ bk,
    const float* __restrict__ Wv, const float* __restrict__ bv,
    const float* __restrict__ Wo, const float* __restrict__ bo,
    float* __restrict__ out) {
  __shared__ Smem smem;
  const int tid = threadIdx.x;

  // ---- stage weights fp32 -> bf16 into LDS (once per workgroup) ----
  for (int i = tid; i < kDim * kDim; i += 256) {
    const int r = i >> 7, c = i & 127;
    smem.wq[r * kWPitch + c] = (__bf16)Wq[i];
    smem.wk[r * kWPitch + c] = (__bf16)Wk[i];
    smem.wv[r * kWPitch + c] = (__bf16)Wv[i];
  }
  for (int i = tid; i < kDim * 256; i += 256) {
    const int r = i >> 8, c = i & 255;
    smem.wo[r * kWoPitch + c] = (__bf16)Wo[i];
  }
  __syncthreads();

  const int lane = tid & 31;
  const int wave = tid >> 5;
  const int ln = lane & 15;  // column index inside 16-wide tile
  const int hi = lane >> 4;  // half-wave selector
  __bf16* qa = smem.ws[wave].qa;
  __bf16* kv = smem.ws[wave].kv;

  // per-lane bias values for column n = nt*16 + ln
  float bqv[8], bkv[8], bvv[8], bov[8];
#pragma unroll
  for (int nt = 0; nt < 8; ++nt) {
    bqv[nt] = bq[nt * 16 + ln];
    bkv[nt] = bk[nt * 16 + ln];
    bvv[nt] = bv[nt * 16 + ln];
    bov[nt] = bo[nt * 16 + ln];
  }

  const int wave_g = blockIdx.x * kWaves + wave;

#pragma unroll 1
  for (int it = 0; it < kBpw; ++it) {
    const int b = wave_g * kBpw + it;
    const float* xb = X + (size_t)b * kEnt * kDim;

    // prefetch next batch's input tile (global_prefetch_b8)
    if (it + 1 < kBpw) {
      const float* nx = xb + kEnt * kDim;
#pragma unroll
      for (int j = 0; j < 4; ++j) __builtin_prefetch(nx + lane * 128 + j * 32, 0, 0);
    }

    // ---- load X into WMMA A-fragment register layout (bf16) ----
    v16bf xA[2][4];
#pragma unroll
    for (int mt = 0; mt < 2; ++mt) {
      const float* xr = xb + (mt * 16 + ln) * kDim;
#pragma unroll
      for (int ks = 0; ks < 4; ++ks) {
        const int c0 = ks * 32 + 16 * hi;
        v4f a0 = *(const v4f*)(xr + c0);
        v4f a1 = *(const v4f*)(xr + c0 + 4);
        v4f b0 = *(const v4f*)(xr + c0 + 8);
        v4f b1 = *(const v4f*)(xr + c0 + 12);
        v16bf f;
#pragma unroll
        for (int j = 0; j < 4; ++j) {
          f[j]      = (__bf16)a0[j];
          f[4 + j]  = (__bf16)a1[j];
          f[8 + j]  = (__bf16)b0[j];
          f[12 + j] = (__bf16)b1[j];
        }
        xA[mt][ks] = f;
      }
    }

    // ---- Q = (X_a Wq^T + bq) * 1/sqrt(128) -> LDS [a][d] ----
#pragma unroll
    for (int nt = 0; nt < 8; ++nt) {
      v8f acc = vzero8();
#pragma unroll
      for (int ks = 0; ks < 4; ++ks)
        acc = wmma_bf16(xA[0][ks],
                        ldB(&smem.wq[(nt * 16 + ln) * kWPitch + ks * 32 + 16 * hi]),
                        acc);
#pragma unroll
      for (int r = 0; r < 8; ++r)
        qa[(r + 8 * hi) * kQaPitch + nt * 16 + ln] =
            (__bf16)((acc[r] + bqv[nt]) * 0.0883883476483184f);
    }

    // ---- K = X Wk^T + bk -> LDS [e][d] ----
#pragma unroll
    for (int mt = 0; mt < 2; ++mt)
#pragma unroll
      for (int nt = 0; nt < 8; ++nt) {
        v8f acc = vzero8();
#pragma unroll
        for (int ks = 0; ks < 4; ++ks)
          acc = wmma_bf16(xA[mt][ks],
                          ldB(&smem.wk[(nt * 16 + ln) * kWPitch + ks * 32 + 16 * hi]),
                          acc);
#pragma unroll
        for (int r = 0; r < 8; ++r)
          kv[(mt * 16 + r + 8 * hi) * kKPitch + nt * 16 + ln] =
              (__bf16)(acc[r] + bkv[nt]);
      }

    // ---- S^T = K Q^T : lane holds all 16 of its entity scores for column a=ln ----
    float sc[2][8];
#pragma unroll
    for (int et = 0; et < 2; ++et) {
      v8f acc = vzero8();
#pragma unroll
      for (int ks = 0; ks < 4; ++ks) {
        v16bf afr = ldB(&kv[(et * 16 + ln) * kKPitch + ks * 32 + 16 * hi]);
        v16bf bfr = ldB(&qa[ln * kQaPitch + ks * 32 + 16 * hi]);
        acc = wmma_bf16(afr, bfr, acc);
      }
      const unsigned long long pm = *(const unsigned long long*)(
          pre_mask + ((size_t)b * 16 + ln) * 32 + et * 16 + 8 * hi);
#pragma unroll
      for (int r = 0; r < 8; ++r) {
        float s = acc[r];
        if ((pm >> (8 * r)) & 0xffull) s = -__builtin_inff();
        const int e = et * 16 + 8 * hi + r;
        if (e == ln) s += -1.0e9f;  // diagonal self-mask
        sc[et][r] = s;
      }
    }

    // ---- softmax over e: 16 lane-local values + partner lane (xor 16) ----
    float mx = -__builtin_inff();
#pragma unroll
    for (int et = 0; et < 2; ++et)
#pragma unroll
      for (int r = 0; r < 8; ++r) mx = fmaxf(mx, sc[et][r]);
    mx = fmaxf(mx, __shfl_xor(mx, 16, 32));
    float ssum = 0.f;
#pragma unroll
    for (int et = 0; et < 2; ++et)
#pragma unroll
      for (int r = 0; r < 8; ++r) {
        sc[et][r] = __expf(sc[et][r] - mx);
        ssum += sc[et][r];
      }
    ssum += __shfl_xor(ssum, 16, 32);
    const float inv = 1.f / ssum;

    // normalized weights land directly in the 16x32 bf16 A-fragment layout
    v16bf wA;
#pragma unroll
    for (int r = 0; r < 8; ++r) {
      wA[r]     = (__bf16)(sc[0][r] * inv);
      wA[8 + r] = (__bf16)(sc[1][r] * inv);
    }

    // ---- V = relu(X Wv^T + bv) -> LDS as V^T [d][e] (packed b128 stores) ----
#pragma unroll
    for (int mt = 0; mt < 2; ++mt)
#pragma unroll
      for (int nt = 0; nt < 8; ++nt) {
        v8f acc = vzero8();
#pragma unroll
        for (int ks = 0; ks < 4; ++ks)
          acc = wmma_bf16(xA[mt][ks],
                          ldB(&smem.wv[(nt * 16 + ln) * kWPitch + ks * 32 + 16 * hi]),
                          acc);
        v8bf pk;
#pragma unroll
        for (int r = 0; r < 8; ++r) pk[r] = (__bf16)fmaxf(acc[r] + bvv[nt], 0.f);
        *(v8bf*)&kv[(nt * 16 + ln) * kVtPitch + mt * 16 + 8 * hi] = pk;
      }

    // ---- attended = weights x V (single K=32 WMMA per d-tile) -> LDS [a][d] ----
#pragma unroll
    for (int nt = 0; nt < 8; ++nt) {
      v8f acc = vzero8();
      acc = wmma_bf16(wA, ldA(&kv[(nt * 16 + ln) * kVtPitch + 8 * hi]), acc);
#pragma unroll
      for (int r = 0; r < 8; ++r)
        qa[(r + 8 * hi) * kQaPitch + nt * 16 + ln] = (__bf16)acc[r];
    }

    // ---- out = [X_a, attended] Wo^T + bo, post-masked ----
    const unsigned long long pm8 =
        *(const unsigned long long*)(post_mask + (size_t)b * 16 + 8 * hi);
    float* ob = out + (size_t)b * kAgents * kDim;
#pragma unroll
    for (int nt = 0; nt < 8; ++nt) {
      v8f acc = vzero8();
#pragma unroll
      for (int ks = 0; ks < 4; ++ks)  // K = 0..127 : X_a from registers
        acc = wmma_bf16(xA[0][ks],
                        ldB(&smem.wo[(nt * 16 + ln) * kWoPitch + ks * 32 + 16 * hi]),
                        acc);
#pragma unroll
      for (int ks = 0; ks < 4; ++ks) {  // K = 128..255 : attended from LDS
        v16bf afr = ldB(&qa[ln * kQaPitch + ks * 32 + 16 * hi]);
        acc = wmma_bf16(
            afr, ldB(&smem.wo[(nt * 16 + ln) * kWoPitch + 128 + ks * 32 + 16 * hi]),
            acc);
      }
#pragma unroll
      for (int r = 0; r < 8; ++r) {
        const int a = r + 8 * hi;
        const bool drop = (pm8 >> (8 * r)) & 0xffull;
        ob[a * kDim + nt * 16 + ln] = drop ? 0.f : (acc[r] + bov[nt]);
      }
    }
  }
}

extern "C" void kernel_launch(void* const* d_in, const int* in_sizes, int n_in,
                              void* d_out, int out_size, void* d_ws, size_t ws_size,
                              hipStream_t stream) {
  const float* X            = (const float*)d_in[0];
  const unsigned char* pre  = (const unsigned char*)d_in[1];
  const unsigned char* post = (const unsigned char*)d_in[2];
  const float* Wq = (const float*)d_in[3];
  const float* bq = (const float*)d_in[4];
  const float* Wk = (const float*)d_in[5];
  const float* bk = (const float*)d_in[6];
  const float* Wv = (const float*)d_in[7];
  const float* bv = (const float*)d_in[8];
  const float* Wo = (const float*)d_in[9];
  const float* bo = (const float*)d_in[10];
  entity_attention_kernel<<<kGrid, 256, 0, stream>>>(X, pre, post, Wq, bq, Wk, bk,
                                                     Wv, bv, Wo, bo, (float*)d_out);
  (void)in_sizes; (void)n_in; (void)out_size; (void)d_ws; (void)ws_size;
}
